// LocalAttention_20753281974466
// MI455X (gfx1250) — matrix operand order repaired and checked by reading, not compiled, not run on hardware
//
#include <hip/hip_runtime.h>
#include <hip/hip_bf16.h>

typedef _Float16 v16h __attribute__((ext_vector_type(16)));
typedef _Float16 v8h  __attribute__((ext_vector_type(8)));
typedef _Float16 v4h  __attribute__((ext_vector_type(4)));
typedef float    v8f  __attribute__((ext_vector_type(8)));
typedef float    v4f  __attribute__((ext_vector_type(4)));

#define WIN      128
#define KW       256          // 2*WIN keys visible per window (look_backward=1)
#define DIM      64
#define NTOK     8192
#define NWIN     (NTOK / WIN) // 64
#define PITCH    264          // padded row pitch in halves for Vt / P (264*2B = 528B = 33*16B)
#define WAVES    8

// One workgroup per (batch, window). 8 waves; each wave owns a 16-row Q strip.
__global__ __launch_bounds__(256)
void local_attn_wmma_kernel(const float* __restrict__ q,
                            const float* __restrict__ k,
                            const float* __restrict__ v,
                            float* __restrict__ out)
{
    __shared__ __align__(16) _Float16 sK [KW * DIM];            // 32 KB, [key][d] row-major
    __shared__ __align__(16) _Float16 sVt[DIM * PITCH];         // 33 KB, [d][key] transposed
    __shared__ __align__(16) _Float16 sP [WAVES * 16 * PITCH];  // 66 KB, per-wave P strips

    const int w    = blockIdx.x;
    const int b    = blockIdx.y;
    const int tid  = threadIdx.x;
    const int wave = tid >> 5;
    const int lane = tid & 31;
    const int l16  = lane & 15;
    const int hi   = lane >> 4;

    const size_t base_tok = (size_t)b * NTOK + (size_t)w * WIN; // first token of this window

    // ------- cooperative stage: K -> sK (f16 row-major), V -> sVt (f16 transposed) -------
    #pragma unroll
    for (int it = 0; it < 16; ++it) {
        int e4  = it * 256 + tid;      // float4 index over 256x64 tile
        int row = e4 >> 4;             // key row 0..255 (0..127 = previous window)
        int col = (e4 & 15) << 2;      // d, step 4
        v4f kv, vv;
        if (w == 0 && row < WIN) {
            kv = -1.0f;                // PAD_VALUE applies to both k and v blocks
            vv = -1.0f;
        } else {
            size_t g = (base_tok + (size_t)(row - WIN)) * DIM + col;
            kv = *(const v4f*)(k + g);
            vv = *(const v4f*)(v + g);
        }
        v4h kh;
        kh[0] = (_Float16)kv[0]; kh[1] = (_Float16)kv[1];
        kh[2] = (_Float16)kv[2]; kh[3] = (_Float16)kv[3];
        *(v4h*)(&sK[row * DIM + col]) = kh;
        #pragma unroll
        for (int j = 0; j < 4; ++j)
            sVt[(col + j) * PITCH + row] = (_Float16)vv[j];
    }
    __syncthreads();

    // ------- per-wave Q strip as WMMA A-operands (pre-scaled by 1/sqrt(64)) -------
    // A operand (this code's contiguous k order): lane holds K = {hi*16 .. hi*16+15}
    const int   m0    = wave * 16;
    const float scale = 0.125f;
    v16h aq[2];
    #pragma unroll
    for (int ks = 0; ks < 2; ++ks) {
        const float* qrow = q + (base_tok + (size_t)(m0 + l16)) * DIM + ks * 32 + hi * 16;
        v4f q0 = *(const v4f*)(qrow);
        v4f q1 = *(const v4f*)(qrow + 4);
        v4f q2 = *(const v4f*)(qrow + 8);
        v4f q3 = *(const v4f*)(qrow + 12);
        v16h a;
        #pragma unroll
        for (int j = 0; j < 4; ++j) {
            a[j]      = (_Float16)(q0[j] * scale);
            a[4 + j]  = (_Float16)(q1[j] * scale);
            a[8 + j]  = (_Float16)(q2[j] * scale);
            a[12 + j] = (_Float16)(q3[j] * scale);
        }
        aq[ks] = a;
    }

    // ------- S = (Q*scale) @ K^T : 16 tiles of 16x16, fp32 accumulate -------
    // B operand (same contiguous k order as A): lane holds 16 contiguous K values of column l16
    //   => B = K^T: lane reads sK[key = t*16 + l16][d = ks*32 + hi*16 .. +15] (contiguous)
    v8f acc[16];
    #pragma unroll
    for (int t = 0; t < 16; ++t) acc[t] = 0.0f;
    #pragma unroll
    for (int t = 0; t < 16; ++t) {
        #pragma unroll
        for (int ks = 0; ks < 2; ++ks) {
            const v16h bk = *(const v16h*)(&sK[(t * 16 + l16) * DIM + ks * 32 + hi * 16]);
            acc[t] = __builtin_amdgcn_wmma_f32_16x16x32_f16(
                         false, aq[ks], false, bk, (short)0, acc[t], false, false);
        }
    }

    // ------- row softmax (C layout: VGPR r, lane -> row r+8*hi, col 16t+l16) -------
    // Row data lives across the 16 lanes of each half-wave -> xor-shuffle masks 1,2,4,8.
    float rinv[8];
    #pragma unroll
    for (int r = 0; r < 8; ++r) {
        float mx = -3.402823466e38f;
        #pragma unroll
        for (int t = 0; t < 16; ++t) mx = fmaxf(mx, acc[t][r]);
        #pragma unroll
        for (int s = 1; s < 16; s <<= 1) mx = fmaxf(mx, __shfl_xor(mx, s, 32));
        float sum = 0.0f;
        #pragma unroll
        for (int t = 0; t < 16; ++t) {
            float p = __expf(acc[t][r] - mx);
            acc[t][r] = p;
            sum += p;
        }
        #pragma unroll
        for (int s = 1; s < 16; s <<= 1) sum += __shfl_xor(sum, s, 32);
        rinv[r] = 1.0f / sum;
    }

    // ------- spill P (unnormalized, f16) to per-wave LDS strip (C -> A relayout) -------
    _Float16* pst = &sP[wave * 16 * PITCH];
    #pragma unroll
    for (int t = 0; t < 16; ++t)
        #pragma unroll
        for (int r = 0; r < 8; ++r)
            pst[(r + hi * 8) * PITCH + t * 16 + l16] = (_Float16)acc[t][r];
    // same-wave LDS RAW dependency: compiler orders via DScnt waits, no barrier needed

    // ------- O = P @ V : A = P strip (16x256), B = V via sVt (contiguous columns) -------
    v8f oacc[4];
    #pragma unroll
    for (int nt = 0; nt < 4; ++nt) oacc[nt] = 0.0f;
    #pragma unroll
    for (int ks = 0; ks < 8; ++ks) {
        const int kb = ks * 32 + hi * 16;
        v8h plo = *(const v8h*)(&pst[l16 * PITCH + kb]);
        v8h phi = *(const v8h*)(&pst[l16 * PITCH + kb + 8]);
        v16h ap = __builtin_shufflevector(plo, phi,
                                          0, 1, 2, 3, 4, 5, 6, 7,
                                          8, 9, 10, 11, 12, 13, 14, 15);
        #pragma unroll
        for (int nt = 0; nt < 4; ++nt) {
            const v16h bv =
                *(const v16h*)(&sVt[(nt * 16 + l16) * PITCH + ks * 32 + hi * 16]);
            oacc[nt] = __builtin_amdgcn_wmma_f32_16x16x32_f16(
                           false, ap, false, bv, (short)0, oacc[nt], false, false);
        }
    }

    // ------- normalize + store fp32 output -------
    #pragma unroll
    for (int nt = 0; nt < 4; ++nt)
        #pragma unroll
        for (int r = 0; r < 8; ++r)
            out[(base_tok + (size_t)(m0 + r + hi * 8)) * DIM + nt * 16 + l16] =
                oacc[nt][r] * rinv[r];
}

extern "C" void kernel_launch(void* const* d_in, const int* in_sizes, int n_in,
                              void* d_out, int out_size, void* d_ws, size_t ws_size,
                              hipStream_t stream) {
    const float* q = (const float*)d_in[0];
    const float* k = (const float*)d_in[1];
    const float* v = (const float*)d_in[2];
    float* out = (float*)d_out;
    const int B = in_sizes[0] / (NTOK * DIM);   // 32
    dim3 grid(NWIN, B);
    dim3 block(256);
    local_attn_wmma_kernel<<<grid, block, 0, stream>>>(q, k, v, out);
}
